// BidirLSTMLayer_45226005627003
// MI455X (gfx1250) — compile-verified, matrix operand order checked by reading, not run on hardware
//
#include <hip/hip_runtime.h>

// Bidirectional LSTM, persistent-kernel recurrence with WMMA bf16 GEMMs.
// dims: T=512, B=128, D=256, H=256, gates 4H=1024, fused K = D+H = 512.
//
// 64 persistent WGs x 128 threads (4 waves): dir(2) x batch-quarter(4, 32 rows)
// x h-col group(8, 32 cols). One wave per SIMD32 -> minimal per-step WMMA issue
// latency on the recurrent critical path. Weights live in LDS for all 512 steps;
// cell state lives in accumulator-layout registers; h ping-pongs through a small
// global bf16 buffer with an 8-WG release/acquire barrier per step.

#define TSTEPS 512
#define BATCH  128
#define DIN    256
#define HID    256
#define GATES  1024          // 4*HID
#define KTOT   512           // DIN + HID
#define LDSP   520           // LDS row pitch in bf16 elems (512 + 8 pad, 16B aligned)

typedef __bf16        v16bf __attribute__((ext_vector_type(16)));
typedef float         v8f   __attribute__((ext_vector_type(8)));
typedef unsigned int  u32x4 __attribute__((ext_vector_type(4)));

union Frag16 { u32x4 u[2]; v16bf v; };   // 16 bf16 = 32B, WMMA A/B operand

__device__ __forceinline__ unsigned short f32_to_bf16(float f) {
  unsigned int u = __builtin_bit_cast(unsigned int, f);
  u += 0x7FFFu + ((u >> 16) & 1u);       // round-to-nearest-even
  return (unsigned short)(u >> 16);
}
__device__ __forceinline__ float sigm(float x) { return 1.0f / (1.0f + __expf(-x)); }
__device__ __forceinline__ float tanh_(float x) {
  float xc = fminf(fmaxf(x, -15.0f), 15.0f);
  float e  = __expf(2.0f * xc);
  return (e - 1.0f) / (e + 1.0f);
}

// ---------------- prep kernels (run once per call, fully parallel) ------------

__global__ void k_cvt_x(const float* __restrict__ x, unsigned short* __restrict__ xbf, int n) {
  int i = blockIdx.x * 256 + threadIdx.x;
  if (i < n) xbf[i] = f32_to_bf16(x[i]);
}

// Wdst[n][k] (row-major, K=512): k<256 from W_ih[n][k], else W_hh[n][k-256]
__global__ void k_build_w(const float* __restrict__ Wih, const float* __restrict__ Whh,
                          unsigned short* __restrict__ Wdst) {
  int i = blockIdx.x * 256 + threadIdx.x;
  if (i >= GATES * KTOT) return;
  int n = i >> 9, k = i & (KTOT - 1);
  float v = (k < DIN) ? Wih[n * DIN + k] : Whh[n * HID + (k - DIN)];
  Wdst[i] = f32_to_bf16(v);
}

__global__ void k_prep_small(const float* __restrict__ bih_f, const float* __restrict__ bhh_f,
                             const float* __restrict__ bih_b, const float* __restrict__ bhh_b,
                             const float* __restrict__ h0,
                             float* __restrict__ bias,
                             unsigned short* __restrict__ hbuf,
                             unsigned int* __restrict__ counters) {
  int i = blockIdx.x * 256 + threadIdx.x;
  if (i < GATES)           bias[i] = bih_f[i] + bhh_f[i];
  else if (i < 2 * GATES)  bias[i] = bih_b[i - GATES] + bhh_b[i - GATES];
  if (i < 2 * BATCH * HID) hbuf[i] = f32_to_bf16(h0[i]);   // ping buffer 0 <- h0[dir][b][j]
  if (i < 8 * TSTEPS)      counters[i] = 0u;
}

// ---------------- persistent recurrent kernel --------------------------------
// grid = 64 WGs: wg>>5 = dir, (wg>>3)&3 = batch quarter (32 rows), wg&7 = h-col group
// block = 128 threads = 4 waves; wave w: mt = w&1 (16-row tile), p = w>>1 (16-col pair)
// Each wave owns 4 WMMA accumulators: gates i,f,g,o for (mt, 16 h-cols).

__global__ __launch_bounds__(128)
void lstm_persistent(const unsigned short* __restrict__ xbf,   // [T][B][D] bf16
                     const unsigned short* __restrict__ Wbf,   // [2][1024][512] bf16
                     const float*          __restrict__ bias,  // [2][1024]
                     unsigned short*       __restrict__ hbuf,  // [2 ping][2 dir][B][H] bf16
                     unsigned int*         __restrict__ counters, // [8][T]
                     const float*          __restrict__ c0,    // [2][B][H]
                     float*                __restrict__ out)   // [T][B][2H] ++ hf,cf,hb,cb
{
  extern __shared__ unsigned short Wlds[];   // [128][LDSP] bf16 = 133,120 B

  const int wg    = blockIdx.x;
  const int dir   = wg >> 5;
  const int rem   = wg & 31;
  const int mq    = rem >> 3;                // batch quarter: 32 rows
  const int jgrp  = rem & 7;
  const int j0    = jgrp * 32;
  const int mrow0 = mq * 32;
  const int grp   = dir * 4 + mq;            // 8 barrier groups of 8 WGs

  const int tid  = threadIdx.x;
  const int wave = tid >> 5;
  const int lane = tid & 31;
  const int mt   = wave & 1;
  const int p    = wave >> 1;                // column pair within 32 h-cols
  const int lrow = lane & 15;
  const int half = lane >> 4;

  // ---- stage this WG's 128 gate rows of W into LDS (once; reused 512 steps) ----
  // local row r: gate = r/32, jj = r%32  ->  global gate row gate*256 + j0 + jj
  const unsigned short* wsrc = Wbf + (size_t)dir * GATES * KTOT;
  for (int idx = tid; idx < 128 * 64; idx += 128) {
    int r = idx >> 6, cw = idx & 63;
    int ng = (r >> 5) * 256 + j0 + (r & 31);
    *(u32x4*)&Wlds[r * LDSP + cw * 8] = *(const u32x4*)&wsrc[(size_t)ng * KTOT + cw * 8];
  }
  __syncthreads();

  // per-lane gate column and biases
  const int    jc   = j0 + p * 16 + lrow;        // h column 0..255
  const float* bptr = bias + dir * GATES;
  const float  bi = bptr[0 * HID + jc], bf = bptr[1 * HID + jc];
  const float  bg = bptr[2 * HID + jc], bo = bptr[3 * HID + jc];

  // cell state lives in registers, matching the 16x16 f32 C/D layout:
  // VGPR r -> row m = mbase + r, col jc   (mbase includes the +8*half lane offset)
  const int mbase = mrow0 + mt * 16 + 8 * half;
  float c[8];
  {
    const float* cp = c0 + ((size_t)dir * BATCH + mbase) * HID + jc;
#pragma unroll
    for (int r = 0; r < 8; ++r) c[r] = cp[(size_t)r * HID];
  }

  const int arow = mrow0 + mt * 16 + lrow;       // A-matrix row this lane supplies
  const size_t OUTMAIN = (size_t)TSTEPS * BATCH * 2 * HID;

  for (int s = 0; s < TSTEPS; ++s) {
    const int t = dir ? (TSTEPS - 1 - s) : s;
    const unsigned short* xrow = xbf  + ((size_t)t * BATCH + arow) * DIN;
    const unsigned short* hrow = hbuf + (((size_t)(s & 1) * 2 + dir) * BATCH + arow) * HID;

    v8f a0 = {}, a1 = {}, a2 = {}, a3 = {};      // acc: i, f, g, o tiles

#pragma unroll 4
    for (int kt = 0; kt < 16; ++kt) {
      // A fragment (16x32 bf16): lane holds row arow, two 8-elem K chunks
      const int k0 = (kt & 7) * 32 + 8 * half;
      const unsigned short* src = (kt < 8) ? (xrow + k0) : (hrow + k0);
      Frag16 fa;
      fa.u[0] = *(const u32x4*)(src);
      fa.u[1] = *(const u32x4*)(src + 16);

      // B fragments (32x16 bf16) from the LDS weight slab, rows [nt*16 + lrow]
      const int lk0 = kt * 32 + 8 * half;
      Frag16 b0, b1, b2, b3;
#define LOADB(F, NT) { const unsigned short* wp = &Wlds[((NT) * 16 + lrow) * LDSP + lk0]; \
                       F.u[0] = *(const u32x4*)(wp); F.u[1] = *(const u32x4*)(wp + 16); }
      LOADB(b0, 0 + p)   // i
      LOADB(b1, 2 + p)   // f
      LOADB(b2, 4 + p)   // g
      LOADB(b3, 6 + p)   // o
#undef LOADB

      a0 = __builtin_amdgcn_wmma_f32_16x16x32_bf16(false, fa.v, false, b0.v, (short)0, a0, false, false);
      a1 = __builtin_amdgcn_wmma_f32_16x16x32_bf16(false, fa.v, false, b1.v, (short)0, a1, false, false);
      a2 = __builtin_amdgcn_wmma_f32_16x16x32_bf16(false, fa.v, false, b2.v, (short)0, a2, false, false);
      a3 = __builtin_amdgcn_wmma_f32_16x16x32_bf16(false, fa.v, false, b3.v, (short)0, a3, false, false);
    }

    // prefetch next step's x row while we do the pointwise work
    if (s + 1 < TSTEPS) {
      const int tn = dir ? (TSTEPS - 2 - s) : (s + 1);
      __builtin_prefetch(xbf + ((size_t)tn * BATCH + arow) * DIN, 0, 1);
    }

    // ---- gate nonlinearities + state update, all in accumulator layout ----
    unsigned short* hw = hbuf + (((size_t)((s + 1) & 1) * 2 + dir) * BATCH) * HID;
    float hnew[8];
#pragma unroll
    for (int r = 0; r < 8; ++r) {
      float ig = sigm (a0[r] + bi);
      float fg = sigm (a1[r] + bf);
      float gg = tanh_(a2[r] + bg);
      float og = sigm (a3[r] + bo);
      float cn = fg * c[r] + ig * gg;
      c[r] = cn;
      float hn = og * tanh_(cn);
      hnew[r] = hn;
      out[((size_t)t * BATCH + mbase + r) * (2 * HID) + dir * HID + jc] = hn;
      hw[(size_t)(mbase + r) * HID + jc] = f32_to_bf16(hn);
    }

    if (s == TSTEPS - 1) {                       // final hT / cT
      float* hT = out + OUTMAIN + (size_t)dir * 2 * BATCH * HID;
      float* cT = hT + (size_t)BATCH * HID;
#pragma unroll
      for (int r = 0; r < 8; ++r) {
        hT[(size_t)(mbase + r) * HID + jc] = hnew[r];
        cT[(size_t)(mbase + r) * HID + jc] = c[r];
      }
    }

    // ---- inter-WG barrier over the 8 WGs sharing (dir, batch-quarter) ----
    __threadfence();                             // release our h stores
    __syncthreads();
    if (tid == 0) {
      unsigned int* slot = &counters[grp * TSTEPS + s];
      __hip_atomic_fetch_add(slot, 1u, __ATOMIC_RELEASE, __HIP_MEMORY_SCOPE_AGENT);
      while (__hip_atomic_load(slot, __ATOMIC_ACQUIRE, __HIP_MEMORY_SCOPE_AGENT) < 8u)
        __builtin_amdgcn_s_sleep(1);
    }
    __syncthreads();
    __threadfence();                             // acquire others' h stores
  }
}

// ---------------- launcher ----------------------------------------------------

extern "C" void kernel_launch(void* const* d_in, const int* in_sizes, int n_in,
                              void* d_out, int out_size, void* d_ws, size_t ws_size,
                              hipStream_t stream) {
  const float* x     = (const float*)d_in[0];
  const float* h0    = (const float*)d_in[1];
  const float* c0    = (const float*)d_in[2];
  const float* Wih_f = (const float*)d_in[3];
  const float* Whh_f = (const float*)d_in[4];
  const float* bih_f = (const float*)d_in[5];
  const float* bhh_f = (const float*)d_in[6];
  const float* Wih_b = (const float*)d_in[7];
  const float* Whh_b = (const float*)d_in[8];
  const float* bih_b = (const float*)d_in[9];
  const float* bhh_b = (const float*)d_in[10];
  float* out = (float*)d_out;
  char*  ws  = (char*)d_ws;

  // workspace layout (bytes)
  const size_t NX      = (size_t)TSTEPS * BATCH * DIN;           // 16,777,216 elems
  const size_t OFF_XBF = 0;                                      // bf16 x
  const size_t OFF_W   = OFF_XBF + NX * 2;                       // +33,554,432
  const size_t OFF_B   = OFF_W   + (size_t)2 * GATES * KTOT * 2; // +2,097,152
  const size_t OFF_H   = OFF_B   + (size_t)2 * GATES * 4;        // +8,192
  const size_t OFF_CNT = OFF_H   + (size_t)2 * 2 * BATCH * HID * 2; // +262,144

  unsigned short* xbf  = (unsigned short*)(ws + OFF_XBF);
  unsigned short* Wbf  = (unsigned short*)(ws + OFF_W);
  float*          bias = (float*)         (ws + OFF_B);
  unsigned short* hbuf = (unsigned short*)(ws + OFF_H);
  unsigned int*   cnt  = (unsigned int*)  (ws + OFF_CNT);

  k_cvt_x   <<<(int)((NX + 255) / 256), 256, 0, stream>>>(x, xbf, (int)NX);
  k_build_w <<<(GATES * KTOT + 255) / 256, 256, 0, stream>>>(Wih_f, Whh_f, Wbf);
  k_build_w <<<(GATES * KTOT + 255) / 256, 256, 0, stream>>>(Wih_b, Whh_b, Wbf + (size_t)GATES * KTOT);
  k_prep_small<<<(2 * BATCH * HID + 255) / 256, 256, 0, stream>>>(
      bih_f, bhh_f, bih_b, bhh_b, h0, bias, hbuf, cnt);

  const size_t ldsBytes = (size_t)128 * LDSP * sizeof(unsigned short); // 133,120 B
  lstm_persistent<<<64, 128, ldsBytes, stream>>>(xbf, Wbf, bias, hbuf, cnt, c0, out);
}
